// MVAE_13486197309656
// MI455X (gfx1250) — matrix operand order repaired and checked by reading, not compiled, not run on hardware
//
#include <hip/hip_runtime.h>

typedef _Float16 f16;
typedef __attribute__((ext_vector_type(16))) _Float16 v16h;
typedef __attribute__((ext_vector_type(8)))  _Float16 v8h;
typedef __attribute__((ext_vector_type(8)))  float    v8f;

#define ROWS_PER_WG 64
#define THREADS 128   // 4 wave32, each wave owns 16 rows

__device__ __forceinline__ float eluf(float v) { return v > 0.f ? v : expm1f(v); }

__device__ __forceinline__ v8f zero8() { v8f z = {}; return z; }

__device__ __forceinline__ v8f wmma16(v16h a, v16h b, v8f c) {
  // D = A(16x32 f16) x B(32x16 f16) + C(16x16 f32)
  return __builtin_amdgcn_wmma_f32_16x16x32_f16(false, a, false, b, (short)0, c, false, false);
}

// A fragment (16x32): lane&15 = row, lanes<16 hold K {0..7,16..23}, lanes>=16 hold K {8..15,24..31}
__device__ __forceinline__ v16h fragA(const f16* base, int stride, int lane, int m0, int kbase) {
  const f16* p = base + (m0 + (lane & 15)) * stride + kbase + ((lane & 16) ? 8 : 0);
  v8h lo = *(const v8h*)(p);
  v8h hi = *(const v8h*)(p + 16);
  v16h r;
#pragma unroll
  for (int i = 0; i < 8; ++i) { r[i] = lo[i]; r[i + 8] = hi[i]; }
  return r;
}

// B fragment (32x16): lane&15 = column N; lanes<16 hold K {0..7,16..23}, lanes>=16 hold K {8..15,24..31}.
// Weights are pre-transposed to [N][K] f16 so the 16 K-values are contiguous.
__device__ __forceinline__ v16h fragB(const f16* __restrict__ W, int K, int lane, int n0, int kbase) {
  const f16* p = W + (size_t)(n0 + (lane & 15)) * K + kbase + ((lane & 16) ? 8 : 0);
  v8h lo = *(const v8h*)(p);
  v8h hi = *(const v8h*)(p + 16);
  v16h r;
#pragma unroll
  for (int i = 0; i < 8; ++i) { r[i] = lo[i]; r[i + 8] = hi[i]; }
  return r;
}

// Generic dense layer: out = elu(in @ W + b) (+ residual after elu), f16 in LDS.
__device__ __forceinline__ void wmma_layer(
    const f16* inL, int K, const f16* __restrict__ Wt, const float* __restrict__ bias,
    const f16* resL, f16* outL, int N, int lane, int m0)
{
  const int ktiles = K >> 5;
  const int nlocal = lane & 15;
  const int rowOff = m0 + ((lane & 16) ? 8 : 0);
  for (int half = 0; half < N; half += 192) {
    v8f C[12];
#pragma unroll
    for (int i = 0; i < 12; ++i) C[i] = zero8();
    for (int kt = 0; kt < ktiles; ++kt) {
      v16h A = fragA(inL, K, lane, m0, kt << 5);
#pragma unroll
      for (int n = 0; n < 12; ++n)
        C[n] = wmma16(A, fragB(Wt, K, lane, half + (n << 4), kt << 5), C[n]);
    }
#pragma unroll
    for (int n = 0; n < 12; ++n) {
      const int col = half + (n << 4) + nlocal;
      const float bv = bias[col];
#pragma unroll
      for (int j = 0; j < 8; ++j) {
        const int row = rowOff + j;
        float v = eluf(C[n][j] + bv);
        if (resL) v += (float)resL[row * K + col];
        outL[row * N + col] = (f16)v;
      }
    }
  }
}

// Mixture-of-experts layer: C accumulates sum_e (diag(coeff_e) * A) x B_e over e and k.
__device__ __forceinline__ void wmma_expert(
    const f16* inL, const float* coeffL,
    const f16* __restrict__ Wt_base, int wstride,
    const float* __restrict__ bias_all, int N,
    const f16* zcAdd, f16* hOut,
    float* __restrict__ gOut, int R0,
    int lane, int m0)
{
  const int nlocal = lane & 15;
  const int rowA = m0 + nlocal;                 // row of this lane's A fragment
  const int rowOff = m0 + ((lane & 16) ? 8 : 0);
  for (int half = 0; half < N; half += 192) {
    v8f C[12];
#pragma unroll
    for (int i = 0; i < 12; ++i) C[i] = zero8();
    for (int e = 0; e < 6; ++e) {
      const f16* Wt = Wt_base + (size_t)e * wstride;
      const f16 s = (f16)coeffL[rowA * 8 + e];
      for (int kt = 0; kt < 12; ++kt) {
        v16h A = fragA(inL, 384, lane, m0, kt << 5);
#pragma unroll
        for (int i = 0; i < 16; ++i) A[i] *= s;  // per-row coeff scaling -> v_pk_mul_f16
#pragma unroll
        for (int n = 0; n < 12; ++n)
          C[n] = wmma16(A, fragB(Wt, 384, lane, half + (n << 4), kt << 5), C[n]);
      }
    }
#pragma unroll
    for (int n = 0; n < 12; ++n) {
      const int col = half + (n << 4) + nlocal;
      float bv[6];
#pragma unroll
      for (int e = 0; e < 6; ++e) bv[e] = bias_all[e * N + col];
#pragma unroll
      for (int j = 0; j < 8; ++j) {
        const int row = rowOff + j;
        float mixb = 0.f;
#pragma unroll
        for (int e = 0; e < 6; ++e) mixb = fmaf(coeffL[row * 8 + e], bv[e], mixb);
        const float acc = C[n][j] + mixb;
        if (hOut) {
          hOut[row * 384 + col] = (f16)((float)zcAdd[row * 384 + col] + eluf(acc));
        } else {
          gOut[(size_t)(R0 + row) * N + col] = acc;
        }
      }
    }
  }
}

// fp32 [mats][K][N] -> f16 [mats][N][Kpad] transposed (zero pad K..Kpad)
__global__ void cvt_tr_kernel(f16* __restrict__ dst, const float* __restrict__ src,
                              int mats, int K, int Kpad, int N, int total) {
  int idx = blockIdx.x * blockDim.x + threadIdx.x;
  if (idx >= total) return;
  int k = idx % Kpad;
  int t = idx / Kpad;
  int n = t % N;
  int m = t / N;
  float v = (k < K) ? src[((size_t)m * K + k) * N + n] : 0.f;
  dst[((size_t)m * N + n) * Kpad + k] = (f16)v;
}

__global__ __launch_bounds__(THREADS) void encode_kernel(
    const float* __restrict__ x, const float* __restrict__ cc, const float* __restrict__ eps,
    const f16* __restrict__ e1t, const float* __restrict__ e1b,
    const f16* __restrict__ e2t, const float* __restrict__ e2b,
    const f16* __restrict__ e3t, const float* __restrict__ e3b,
    const f16* __restrict__ e4t, const float* __restrict__ e4b,
    const f16* __restrict__ mulvt, const float* __restrict__ mub, const float* __restrict__ lvb,
    const f16* __restrict__ d1t, const float* __restrict__ d1b,
    f16* __restrict__ zcW, float* __restrict__ muO, float* __restrict__ lvO)
{
  extern __shared__ char smem[];
  f16*   bufA  = (f16*)smem;                                  // 64*384 f16
  f16*   bufB  = bufA + ROWS_PER_WG * 384;                    // 64*384 f16
  float* muBuf = (float*)(bufB + ROWS_PER_WG * 384);          // 64*40 f32
  float* lvBuf = muBuf + ROWS_PER_WG * 40;                    // 64*40 f32
  f16*   zpad  = (f16*)(lvBuf + ROWS_PER_WG * 40);            // 64*64 f16 (K padded 40->64)

  const int tid = threadIdx.x;
  const int lane = tid & 31;
  const int m0 = (tid >> 5) * 16;
  const int R0 = blockIdx.x * ROWS_PER_WG;

  // stage xc = [x|c] as f16
  for (int i = tid; i < ROWS_PER_WG * 384; i += THREADS) {
    int row = i / 384, col = i - row * 384;
    float v = (col < 192) ? x[(size_t)(R0 + row) * 192 + col]
                          : cc[(size_t)(R0 + row) * 192 + (col - 192)];
    bufA[i] = (f16)v;
  }
  __syncthreads();
  wmma_layer(bufA, 384, e1t, e1b, bufA, bufB, 384, lane, m0);      // h_in = xc + elu(xc W1 + b1)
  __syncthreads();
  wmma_layer(bufB, 384, e2t, e2b, nullptr, bufA, 384, lane, m0);   // h2
  __syncthreads();
  wmma_layer(bufA, 384, e3t, e3b, bufA, bufB, 384, lane, m0);      // s = h2 + h3
  __syncthreads();
  wmma_layer(bufB, 384, e4t, e4b, nullptr, bufA, 192, lane, m0);   // h4 (f16, [64][192])
  __syncthreads();

  // mu / logvar : h4 @ [muw|lvw]  (N = 80, f32 accumulate)
  {
    const int nlocal = lane & 15;
    const int rowOff = m0 + ((lane & 16) ? 8 : 0);
    v8f C[5];
#pragma unroll
    for (int i = 0; i < 5; ++i) C[i] = zero8();
    for (int kt = 0; kt < 6; ++kt) {
      v16h A = fragA(bufA, 192, lane, m0, kt << 5);
#pragma unroll
      for (int n = 0; n < 5; ++n)
        C[n] = wmma16(A, fragB(mulvt, 192, lane, n << 4, kt << 5), C[n]);
    }
#pragma unroll
    for (int n = 0; n < 5; ++n) {
      const int col = (n << 4) + nlocal;
#pragma unroll
      for (int j = 0; j < 8; ++j) {
        const int row = rowOff + j;
        float v = C[n][j];
        if (col < 40) {
          v += mub[col];
          muBuf[row * 40 + col] = v;
          muO[(size_t)(R0 + row) * 40 + col] = v;
        } else {
          const int lj = col - 40;
          v = fminf(v + lvb[lj], 20.f);
          lvBuf[row * 40 + lj] = v;
          lvO[(size_t)(R0 + row) * 40 + lj] = v;
        }
      }
    }
  }
  __syncthreads();
  // z = mu + eps * exp(0.5*logvar), zero-padded to K=64 for the decoder GEMM
  for (int i = tid; i < ROWS_PER_WG * 64; i += THREADS) {
    int row = i >> 6, j = i & 63;
    f16 v = (f16)0.f;
    if (j < 40) {
      float z = muBuf[row * 40 + j] +
                eps[(size_t)(R0 + row) * 40 + j] * expf(0.5f * lvBuf[row * 40 + j]);
      v = (f16)z;
    }
    zpad[i] = v;
  }
  __syncthreads();
  // zdec = elu(z @ d1w + d1b) -> zc[:,0:192] ; copy c -> zc[:,192:384]
  {
    const int nlocal = lane & 15;
    const int rowOff = m0 + ((lane & 16) ? 8 : 0);
    v8f C[12];
#pragma unroll
    for (int i = 0; i < 12; ++i) C[i] = zero8();
    for (int kt = 0; kt < 2; ++kt) {
      v16h A = fragA(zpad, 64, lane, m0, kt << 5);
#pragma unroll
      for (int n = 0; n < 12; ++n)
        C[n] = wmma16(A, fragB(d1t, 64, lane, n << 4, kt << 5), C[n]);
    }
#pragma unroll
    for (int n = 0; n < 12; ++n) {
      const int col = (n << 4) + nlocal;
#pragma unroll
      for (int j = 0; j < 8; ++j) {
        const int row = rowOff + j;
        zcW[(size_t)(R0 + row) * 384 + col] = (f16)eluf(C[n][j] + d1b[col]);
      }
    }
  }
  for (int i = tid; i < ROWS_PER_WG * 192; i += THREADS) {
    int row = i / 192, col = i - row * 192;
    zcW[(size_t)(R0 + row) * 384 + 192 + col] = (f16)cc[(size_t)(R0 + row) * 192 + col];
  }
}

__global__ __launch_bounds__(THREADS) void moe_kernel(
    const f16* __restrict__ zcW,
    const f16* __restrict__ g1t, const float* __restrict__ g1b,
    const float* __restrict__ g2w, const float* __restrict__ g2b,
    const float* __restrict__ g3w, const float* __restrict__ g3b,
    const f16* __restrict__ w0t, const float* __restrict__ b0,
    const f16* __restrict__ w1t, const float* __restrict__ b1,
    float* __restrict__ outO)
{
  extern __shared__ char smem[];
  f16*   zcL    = (f16*)smem;                                 // 64*384 f16
  f16*   hL     = zcL + ROWS_PER_WG * 384;                    // 64*384 f16 (zc + h)
  float* gbuf   = (float*)(hL + ROWS_PER_WG * 384);           // 64*64 f32
  float* g2buf  = gbuf + ROWS_PER_WG * 64;                    // 64*64 f32
  float* coeffL = g2buf + ROWS_PER_WG * 64;                   // 64*8 f32

  const int tid = threadIdx.x;
  const int lane = tid & 31;
  const int m0 = (tid >> 5) * 16;
  const int R0 = blockIdx.x * ROWS_PER_WG;

  {
    const int4* src = (const int4*)(zcW + (size_t)R0 * 384);
    int4* dst = (int4*)zcL;
    for (int i = tid; i < ROWS_PER_WG * 384 * 2 / 16; i += THREADS) dst[i] = src[i];
  }
  __syncthreads();
  // g1 = elu(zc @ g1w + g1b)  (N = 64)
  {
    const int nlocal = lane & 15;
    const int rowOff = m0 + ((lane & 16) ? 8 : 0);
    v8f C[4];
#pragma unroll
    for (int i = 0; i < 4; ++i) C[i] = zero8();
    for (int kt = 0; kt < 12; ++kt) {
      v16h A = fragA(zcL, 384, lane, m0, kt << 5);
#pragma unroll
      for (int n = 0; n < 4; ++n)
        C[n] = wmma16(A, fragB(g1t, 384, lane, n << 4, kt << 5), C[n]);
    }
#pragma unroll
    for (int n = 0; n < 4; ++n) {
      const int col = (n << 4) + nlocal;
#pragma unroll
      for (int j = 0; j < 8; ++j)
        gbuf[(rowOff + j) * 64 + col] = eluf(C[n][j] + g1b[col]);
    }
  }
  __syncthreads();
  // g2 = elu(g @ g2w + g2b)  (tiny, VALU fp32; 2 threads per row)
  {
    const int row = tid >> 1, part = tid & 1;
    for (int j = part * 32; j < part * 32 + 32; ++j) {
      float acc = g2b[j];
      for (int k = 0; k < 64; ++k) acc = fmaf(gbuf[row * 64 + k], g2w[k * 64 + j], acc);
      g2buf[row * 64 + j] = eluf(acc);
    }
  }
  __syncthreads();
  // coeff = softmax(g2 @ g3w + g3b)
  if ((tid & 1) == 0) {
    const int row = tid >> 1;
    float l[6];
    float mx = -1e30f;
#pragma unroll
    for (int j = 0; j < 6; ++j) {
      float acc = g3b[j];
      for (int k = 0; k < 64; ++k) acc = fmaf(g2buf[row * 64 + k], g3w[k * 6 + j], acc);
      l[j] = acc;
      mx = fmaxf(mx, acc);
    }
    float s = 0.f;
#pragma unroll
    for (int j = 0; j < 6; ++j) { l[j] = expf(l[j] - mx); s += l[j]; }
    const float inv = 1.f / s;
#pragma unroll
    for (int j = 0; j < 6; ++j) coeffL[row * 8 + j] = l[j] * inv;
    coeffL[row * 8 + 6] = 0.f;
    coeffL[row * 8 + 7] = 0.f;
  }
  __syncthreads();
  // expert layer 0: hL = zc + elu(mix(zc @ w0) + coeff@b0)
  wmma_expert(zcL, coeffL, w0t, 384 * 384, b0, 384, zcL, hL, nullptr, 0, lane, m0);
  __syncthreads();
  // expert layer 1: out = mix((zc+h) @ w1) + coeff@b1
  wmma_expert(hL, coeffL, w1t, 192 * 384, b1, 192, nullptr, nullptr, outO, R0, lane, m0);
}

extern "C" void kernel_launch(void* const* d_in, const int* in_sizes, int n_in,
                              void* d_out, int out_size, void* d_ws, size_t ws_size,
                              hipStream_t stream) {
  (void)n_in; (void)out_size; (void)ws_size;
  const float* x   = (const float*)d_in[0];
  const float* c   = (const float*)d_in[1];
  const float* eps = (const float*)d_in[2];
  const float* e1w = (const float*)d_in[3];  const float* e1b = (const float*)d_in[4];
  const float* e2w = (const float*)d_in[5];  const float* e2b = (const float*)d_in[6];
  const float* e3w = (const float*)d_in[7];  const float* e3b = (const float*)d_in[8];
  const float* e4w = (const float*)d_in[9];  const float* e4b = (const float*)d_in[10];
  const float* muw = (const float*)d_in[11]; const float* mub = (const float*)d_in[12];
  const float* lvw = (const float*)d_in[13]; const float* lvb = (const float*)d_in[14];
  const float* d1w = (const float*)d_in[15]; const float* d1b = (const float*)d_in[16];
  const float* w0  = (const float*)d_in[17]; const float* b0  = (const float*)d_in[18];
  const float* w1  = (const float*)d_in[19]; const float* b1  = (const float*)d_in[20];
  const float* g1w = (const float*)d_in[21]; const float* g1b = (const float*)d_in[22];
  const float* g2w = (const float*)d_in[23]; const float* g2b = (const float*)d_in[24];
  const float* g3w = (const float*)d_in[25]; const float* g3b = (const float*)d_in[26];

  const int rows = in_sizes[0] / 192;  // B
  float* outO = (float*)d_out;
  float* muO  = outO + (size_t)rows * 192;
  float* lvO  = muO + (size_t)rows * 40;

  char* ws = (char*)d_ws;
  size_t off = 0;
  auto take = [&](size_t elems) {
    f16* p = (f16*)(ws + off);
    off = (off + elems * sizeof(f16) + 255) & ~(size_t)255;
    return p;
  };
  f16* e1t   = take((size_t)384 * 384);
  f16* e2t   = take((size_t)384 * 384);
  f16* e3t   = take((size_t)384 * 384);
  f16* e4t   = take((size_t)192 * 384);
  f16* g1t   = take((size_t)64 * 384);
  f16* mulvt = take((size_t)80 * 192);
  f16* d1t   = take((size_t)192 * 64);
  f16* w0t   = take((size_t)6 * 384 * 384);
  f16* w1t   = take((size_t)6 * 192 * 384);
  f16* zcW   = take((size_t)rows * 384);

  auto cvt = [&](f16* dst, const float* src, int mats, int K, int Kpad, int N) {
    int total = mats * N * Kpad;
    cvt_tr_kernel<<<(total + 255) / 256, 256, 0, stream>>>(dst, src, mats, K, Kpad, N, total);
  };
  cvt(e1t, e1w, 1, 384, 384, 384);
  cvt(e2t, e2w, 1, 384, 384, 384);
  cvt(e3t, e3w, 1, 384, 384, 384);
  cvt(e4t, e4w, 1, 384, 384, 192);
  cvt(g1t, g1w, 1, 384, 384, 64);
  cvt(mulvt, muw, 1, 192, 192, 40);
  cvt(mulvt + 40 * 192, lvw, 1, 192, 192, 40);
  cvt(d1t, d1w, 1, 40, 64, 192);
  cvt(w0t, w0, 6, 384, 384, 384);
  cvt(w1t, w1, 6, 384, 384, 192);

  const int nwg = rows / ROWS_PER_WG;
  const size_t sh1 = (size_t)ROWS_PER_WG * 384 * 2 * 2   // bufA + bufB
                   + (size_t)ROWS_PER_WG * 40 * 4 * 2    // muBuf + lvBuf
                   + (size_t)ROWS_PER_WG * 64 * 2;       // zpad
  encode_kernel<<<nwg, THREADS, sh1, stream>>>(
      x, c, eps, e1t, e1b, e2t, e2b, e3t, e3b, e4t, e4b,
      mulvt, mub, lvb, d1t, d1b, zcW, muO, lvO);

  const size_t sh2 = (size_t)ROWS_PER_WG * 384 * 2 * 2   // zcL + hL
                   + (size_t)ROWS_PER_WG * 64 * 4 * 2    // gbuf + g2buf
                   + (size_t)ROWS_PER_WG * 8 * 4;        // coeff
  moe_kernel<<<nwg, THREADS, sh2, stream>>>(
      zcW, g1t, g1b, g2w, g2b, g3w, g3b, w0t, b0, w1t, b1, outO);
}
